// TemporalAttention_65601330479408
// MI455X (gfx1250) — matrix operand order repaired and checked by reading, not compiled, not run on hardware
//
#include <hip/hip_runtime.h>

#define N 8192
#define T 12
#define KSPLIT 4

typedef __attribute__((ext_vector_type(16))) _Float16 v16h;
typedef __attribute__((ext_vector_type(8)))  _Float16 v8h;
typedef __attribute__((ext_vector_type(8)))  float    v8f;
typedef __attribute__((ext_vector_type(4)))  float    v4f;

// exp(s/sqrt(8192)) = exp2(s * log2(e)/sqrt(8192)); fold sqrt of that scale
// into BOTH WMMA1 operands (they multiply) so the gemm output feeds exp2 raw.
#define QSCALE 0.12625468f   // sqrt(1.44269504 / 90.50966799)

// ---------------------------------------------------------------------------
// Pass 1: build the f16 operand buffers.
//   qaA[N][32] : per key row, the A-operand image pre-padded to the WMMA
//                register layout: [t0..t7, 0 x8, t8..t15, 0 x8] (scaled).
//                Lane loads 32B at element offset half*16 -> operand tuple
//                directly, no register splicing.
//   vtp[16][N] : V^T rows, UNscaled, keys stored in the WMMA-slot-permuted
//                order; row t=12 is all ones (exp-sum via tensor core),
//                rows 13..15 zero.
// Slot order per 32-key group: local key r = 16p + 8hi + v maps to
// slot s = 2v + p + 16hi, stored at pos = s
// (so each lane's V^T A-operand load is 32B contiguous).
// ---------------------------------------------------------------------------
__global__ __launch_bounds__(256) void prep_kernel(const float* __restrict__ q,
                                                   _Float16* __restrict__ qaA,
                                                   _Float16* __restrict__ vtp) {
  const int n = blockIdx.x * 256 + threadIdx.x;
  if (n >= N) return;

  float r[T];
#pragma unroll
  for (int t = 0; t < T; ++t) r[t] = q[n * T + t];

  v16h lo = {};   // [t0..t7, 0 x8]
  v16h hi = {};   // [t8..t11, 0 x12]
#pragma unroll
  for (int t = 0; t < 8; ++t) lo[t] = (_Float16)(r[t] * QSCALE);
#pragma unroll
  for (int t = 8; t < T; ++t) hi[t - 8] = (_Float16)(r[t] * QSCALE);
  *(v16h*)(qaA + n * 32)      = lo;              // 32B contiguous
  *(v16h*)(qaA + n * 32 + 16) = hi;

  // permuted position of key n inside its 32-key group
  const int g   = n >> 5;
  const int rr  = n & 31;
  const int p   = (rr >> 4) & 1;
  const int h8  = (rr >> 3) & 1;
  const int v   = rr & 7;
  const int s   = 2 * v + p + 16 * h8;
  const int pos = s;
  const int col = g * 32 + pos;

#pragma unroll
  for (int t = 0; t < T; ++t) vtp[t * N + col] = (_Float16)r[t];
  vtp[12 * N + col] = (_Float16)1.0f;            // ones row -> free exp-sum
  vtp[13 * N + col] = (_Float16)0.0f;
  vtp[14 * N + col] = (_Float16)0.0f;
  vtp[15 * N + col] = (_Float16)0.0f;
}

// ---------------------------------------------------------------------------
// Pass 2: fused attention, one wave per (q-tile, k-split), 32 keys/iter.
//   S^T tiles  : 2x v_wmma_f32_16x16x32_f16 (K=12 padded)
//   P^T        : exp2(S^T) (scale pre-folded; |s|<~0.5 so no max needed)
//   O^T        : 1x v_wmma_f32_16x16x32_f16, full K=32 (permuted key slots)
// Row t=12 of O^T accumulates the exp-sum (ones row of V^T).
// ---------------------------------------------------------------------------
__global__ __launch_bounds__(256) void fa_kernel(const _Float16* __restrict__ qaA,
                                                 const _Float16* __restrict__ vtp,
                                                 float* __restrict__ po) {
  const int wave = blockIdx.x * 8 + (threadIdx.x >> 5);   // 0..2047
  const int lane = threadIdx.x & 31;
  const int half = lane >> 4;                              // 0 | 1
  const int l15  = lane & 15;
  const int qt   = wave >> 2;                              // q tile 0..511
  const int ks   = wave & (KSPLIT - 1);                    // k split
  const int q0   = qt * 16;

  // Loop-invariant B operand (Q^T 32x16): lanes 0-15 hold t=0..7, lanes 16-31
  // hold t=8..15 in elements 0..7; elements 8..15 (t=16..31 padding) are zero.
  // Assembled once from the padded buffer (t8..15 live at element 16..23).
  v16h bq;
  {
    v8h qlo = *(const v8h*)(qaA + (q0 + l15) * 32);
    v8h qhi = *(const v8h*)(qaA + (q0 + l15) * 32 + 16);
#pragma unroll
    for (int j = 0; j < 8; ++j) {
      bq[j]     = half ? qhi[j] : qlo[j];
      bq[j + 8] = (_Float16)0.0f;
    }
  }

  v8f o = {};   // O^T accumulator: (t = reg + 8*half, q = l15)

  const int kbeg = ks * (N / KSPLIT);
  const int kend = kbeg + (N / KSPLIT);
  for (int k0 = kbeg; k0 < kend; k0 += 32) {
    // A operands: two 16-key tiles, pre-padded in memory -> direct 32B loads.
    v16h ak0 = *(const v16h*)(qaA + (k0 + l15) * 32 + half * 16);
    v16h ak1 = *(const v16h*)(qaA + (k0 + 16 + l15) * 32 + half * 16);
    __builtin_prefetch(qaA + (k0 + 32 + l15) * 32, 0, 3);

    // S^T tiles: reg v, lane l -> (key = v + 8*half (+16), query = l15)
    v8f s0 = {}, s1 = {};
    s0 = __builtin_amdgcn_wmma_f32_16x16x32_f16(false, ak0, false, bq,
                                                (short)0, s0, false, false);
    s1 = __builtin_amdgcn_wmma_f32_16x16x32_f16(false, ak1, false, bq,
                                                (short)0, s1, false, false);

    // P^T = exp2(S^T); pack straight into the B-operand slot layout:
    // element 2v <- tile0 reg v, element 2v+1 <- tile1 reg v. No shuffles.
    v16h bp;
#pragma unroll
    for (int v = 0; v < 8; ++v) {
      bp[2 * v]     = (_Float16)__builtin_amdgcn_exp2f(s0[v]);
      bp[2 * v + 1] = (_Float16)__builtin_amdgcn_exp2f(s1[v]);
    }

    // A operand: V^T row t=l15, 16 permuted key slots, 32B contiguous.
    v16h av = *(const v16h*)(vtp + l15 * N + k0 + half * 16);

    o = __builtin_amdgcn_wmma_f32_16x16x32_f16(false, av, false, bp,
                                               (short)0, o, false, false);
  }

  // Store additive partials, coalesced: 2 x b128 per lane.
  v4f o0 = {o[0], o[1], o[2], o[3]};
  v4f o1 = {o[4], o[5], o[6], o[7]};
  *(v4f*)(po + wave * 256 + lane * 8)     = o0;
  *(v4f*)(po + wave * 256 + lane * 8 + 4) = o1;
}

// ---------------------------------------------------------------------------
// Pass 3: sum the KSPLIT additive partials, normalize by the exp-sum that
// the tensor core accumulated in row t=12, write O (N x 12, f32).
// ---------------------------------------------------------------------------
__global__ __launch_bounds__(256) void combine_kernel(const float* __restrict__ po,
                                                      float* __restrict__ out) {
  const int w    = blockIdx.x * 8 + (threadIdx.x >> 5);   // q tile 0..511
  const int lane = threadIdx.x & 31;
  const int half = lane >> 4;
  const int l15  = lane & 15;

  v4f a0 = {}; v4f a1 = {};
#pragma unroll
  for (int ks = 0; ks < KSPLIT; ++ks) {
    const int wv = w * KSPLIT + ks;
    a0 += *(const v4f*)(po + wv * 256 + lane * 8);
    a1 += *(const v4f*)(po + wv * 256 + lane * 8 + 4);
  }
  // exp-sum lives at t=12 -> (half==1, reg 4) == a1[0] of lanes 16..31.
  const float partner = __shfl_xor(a1[0], 16, 32);
  const float es = half ? a1[0] : partner;
  const float inv = 1.0f / es;
  a0 *= inv;
  a1 *= inv;

  const int q = w * 16 + l15;
  float* base = out + q * T;          // q*48 bytes -> 16B aligned
  if (!half) {
    *(v4f*)(base + 0) = a0;           // t = 0..3
    *(v4f*)(base + 4) = a1;           // t = 4..7
  } else {
    *(v4f*)(base + 8) = a0;           // t = 8..11 (t>=12 padding dropped)
  }
}

// ---------------------------------------------------------------------------
extern "C" void kernel_launch(void* const* d_in, const int* in_sizes, int n_in,
                              void* d_out, int out_size, void* d_ws, size_t ws_size,
                              hipStream_t stream) {
  const float* q = (const float*)d_in[0];   // queries (values input unused, per reference)
  float* out = (float*)d_out;

  char* ws = (char*)d_ws;
  _Float16* qaA = (_Float16*)ws;                                  // 512 KB
  _Float16* vtp = (_Float16*)(ws + 512 * 1024);                   // 256 KB
  float*    po  = (float*)   (ws + 768 * 1024);                   // 2 MB

  prep_kernel   <<<N / 256,            256, 0, stream>>>(q, qaA, vtp);
  fa_kernel     <<<(512 * KSPLIT) / 8, 256, 0, stream>>>(qaA, vtp, po);
  combine_kernel<<<512 / 8,            256, 0, stream>>>(po, out);
}
